// DecoderGRU_4337916969955
// MI455X (gfx1250) — matrix operand order repaired and checked by reading, not compiled, not run on hardware
//
#include <hip/hip_runtime.h>
#include <hip/hip_bf16.h>
#include <math.h>

typedef __bf16 bf16_t;
typedef __attribute__((ext_vector_type(16))) __bf16 v16bf;
typedef __attribute__((ext_vector_type(8)))  __bf16 v8bf;
typedef __attribute__((ext_vector_type(8)))  float  v8f;

#define HID   768
#define VOCAB 32000
#define BATCH 32
#define SEQ   64
#define G3    (3*HID)      /* 2304 */
#define XDIM  (2*HID)      /* 1536 */

static __device__ __forceinline__ float sigmoidf_(float x) {
    return 1.0f / (1.0f + __expf(-x));
}

// ---------------------------------------------------------------------------
// Elementwise f32 -> bf16 conversion
// ---------------------------------------------------------------------------
__global__ void f32_to_bf16_kernel(const float* __restrict__ src,
                                   bf16_t* __restrict__ dst, size_t n) {
    size_t i = (size_t)blockIdx.x * blockDim.x + threadIdx.x;
    if (i < n) dst[i] = (bf16_t)src[i];
}

// Gather embedding rows in (t,b) row order, convert to bf16
__global__ void gather_emb_kernel(const int* __restrict__ tok,
                                  const float* __restrict__ emb,
                                  bf16_t* __restrict__ dst) {
    size_t i = (size_t)blockIdx.x * blockDim.x + threadIdx.x; // 2048*768
    if (i >= (size_t)BATCH * SEQ * HID) return;
    int k = (int)(i % HID);
    int r = (int)(i / HID);
    int t = r >> 5;      // r / 32
    int b = r & 31;
    int tk = tok[b * SEQ + t];
    dst[i] = (bf16_t)emb[(size_t)tk * HID + k];
}

__global__ void zero_counter_kernel(unsigned* __restrict__ c) {
    if (threadIdx.x == 0 && blockIdx.x == 0) c[0] = 0u;
}

// ---------------------------------------------------------------------------
// Fragment load helpers (contiguous k order, same for A and B; not the ISA table order)
// A (16x32, M x K): lane = half*16 + r16 holds row r16; elements e hold
// K = kb + 16*half + e (two 16B loads).
// B (N x K row-major, used as K x N): lane holds column n = r16;
// elements e hold K = kb + 16*half + e (one contiguous 32B load).
// ---------------------------------------------------------------------------
static __device__ __forceinline__ v16bf load_a_frag(const bf16_t* Arow, int kb, int half) {
    v8bf a0 = *reinterpret_cast<const v8bf*>(Arow + kb + 16 * half);
    v8bf a1 = *reinterpret_cast<const v8bf*>(Arow + kb + 16 * half + 8);
    v16bf af;
#pragma unroll
    for (int i = 0; i < 8; ++i) { af[i] = a0[i]; af[i + 8] = a1[i]; }
    return af;
}

// ---------------------------------------------------------------------------
// Register-blocked GEMM: C(M,N) = A(M,K) @ B(N,K)^T (+bias).
// 256 threads = 8 waves; wave grid = 4(M) x 2(N); each wave computes a
// 32(M) x 64(N) strip with 8 accumulators: per K-step 2 A frags + 4 B frags
// feed 8 WMMA (12 b128 loads / 8 wmma = 1.5 loads per wmma).
// WG tile = 128(M) x 128(N).  grid = (N/128, M/128).
// permute_ts: C row gm=(t*32+b) stored at out[(b*SEQ+t)*ldc + gn].
// ---------------------------------------------------------------------------
__global__ void __launch_bounds__(256)
gemm_bf16_tn_blk_kernel(const bf16_t* __restrict__ A, int lda,
                        const bf16_t* __restrict__ B, int ldb,
                        const float* __restrict__ bias,
                        float* __restrict__ C, int ldc,
                        int K, int permute_ts) {
    const int lane = threadIdx.x & 31;
    const int wave = threadIdx.x >> 5;
    const int half = lane >> 4;
    const int r16  = lane & 15;
    const int msub = wave >> 1;      // 0..3
    const int ngrp = wave & 1;       // 0..1
    const int m0     = blockIdx.y * 128 + msub * 32;
    const int n_base = blockIdx.x * 128 + ngrp * 64;

    const bf16_t* Arow0 = A + (size_t)(m0 + r16) * lda;
    const bf16_t* Arow1 = A + (size_t)(m0 + 16 + r16) * lda;
    const bf16_t* Brow0 = B + (size_t)(n_base +  0 + r16) * ldb;
    const bf16_t* Brow1 = B + (size_t)(n_base + 16 + r16) * ldb;
    const bf16_t* Brow2 = B + (size_t)(n_base + 32 + r16) * ldb;
    const bf16_t* Brow3 = B + (size_t)(n_base + 48 + r16) * ldb;

    v8f acc00 = {}, acc01 = {}, acc02 = {}, acc03 = {};
    v8f acc10 = {}, acc11 = {}, acc12 = {}, acc13 = {};
    for (int kb = 0; kb < K; kb += 32) {
        v16bf af0 = load_a_frag(Arow0, kb, half);
        v16bf af1 = load_a_frag(Arow1, kb, half);
        v16bf b0 = *reinterpret_cast<const v16bf*>(Brow0 + kb + 16 * half);
        v16bf b1 = *reinterpret_cast<const v16bf*>(Brow1 + kb + 16 * half);
        v16bf b2 = *reinterpret_cast<const v16bf*>(Brow2 + kb + 16 * half);
        v16bf b3 = *reinterpret_cast<const v16bf*>(Brow3 + kb + 16 * half);
        acc00 = __builtin_amdgcn_wmma_f32_16x16x32_bf16(false, af0, false, b0, (short)0, acc00, false, false);
        acc01 = __builtin_amdgcn_wmma_f32_16x16x32_bf16(false, af0, false, b1, (short)0, acc01, false, false);
        acc02 = __builtin_amdgcn_wmma_f32_16x16x32_bf16(false, af0, false, b2, (short)0, acc02, false, false);
        acc03 = __builtin_amdgcn_wmma_f32_16x16x32_bf16(false, af0, false, b3, (short)0, acc03, false, false);
        acc10 = __builtin_amdgcn_wmma_f32_16x16x32_bf16(false, af1, false, b0, (short)0, acc10, false, false);
        acc11 = __builtin_amdgcn_wmma_f32_16x16x32_bf16(false, af1, false, b1, (short)0, acc11, false, false);
        acc12 = __builtin_amdgcn_wmma_f32_16x16x32_bf16(false, af1, false, b2, (short)0, acc12, false, false);
        acc13 = __builtin_amdgcn_wmma_f32_16x16x32_bf16(false, af1, false, b3, (short)0, acc13, false, false);
    }

    const v8f* accs[2][4] = { { &acc00, &acc01, &acc02, &acc03 },
                              { &acc10, &acc11, &acc12, &acc13 } };
#pragma unroll
    for (int mi = 0; mi < 2; ++mi) {
        const int gm = m0 + mi * 16 + half * 8;   // rows gm..gm+7 (element v)
        size_t rowoff[8];
#pragma unroll
        for (int v = 0; v < 8; ++v) {
            int gmr = gm + v;
            if (permute_ts) {
                int t = gmr >> 5, b = gmr & 31;
                rowoff[v] = ((size_t)b * SEQ + t) * (size_t)ldc;
            } else {
                rowoff[v] = (size_t)gmr * ldc;
            }
        }
#pragma unroll
        for (int j = 0; j < 4; ++j) {
            int gn = n_base + j * 16 + r16;
            float bv = bias ? bias[gn] : 0.0f;
#pragma unroll
            for (int v = 0; v < 8; ++v)
                C[rowoff[v] + gn] = (*accs[mi][j])[v] + bv;
        }
    }
}

// ---------------------------------------------------------------------------
// Simple GEMM (1 tile / wave) kept for the tiny M=32 gimg GEMM.
// WG tile = 32(M) x 64(N); grid = (N/64, M/32).
// ---------------------------------------------------------------------------
__global__ void __launch_bounds__(256)
gemm_bf16_tn_kernel(const bf16_t* __restrict__ A, int lda,
                    const bf16_t* __restrict__ B, int ldb,
                    const float* __restrict__ bias,
                    float* __restrict__ C, int ldc,
                    int K) {
    const int lane = threadIdx.x & 31;
    const int wave = threadIdx.x >> 5;
    const int half = lane >> 4;
    const int r16  = lane & 15;
    const int m0 = blockIdx.y * 32 + (wave >> 2) * 16;
    const int n0 = blockIdx.x * 64 + (wave & 3) * 16;

    const bf16_t* Arow = A + (size_t)(m0 + r16) * lda;
    const bf16_t* Brow = B + (size_t)(n0 + r16) * ldb;

    v8f acc = {};
    for (int kb = 0; kb < K; kb += 32) {
        v16bf af = load_a_frag(Arow, kb, half);
        v16bf bfrag = *reinterpret_cast<const v16bf*>(Brow + kb + 16 * half);
        acc = __builtin_amdgcn_wmma_f32_16x16x32_bf16(
            false, af, false, bfrag, (short)0, acc, false, false);
    }
#pragma unroll
    for (int v = 0; v < 8; ++v) {
        int gm = m0 + half * 8 + v;
        int gn = n0 + r16;
        C[(size_t)gm * ldc + gn] = acc[v] + (bias ? bias[gn] : 0.0f);
    }
}

// ---------------------------------------------------------------------------
// Persistent GRU recurrence: 48 WGs x 256 threads, WG owns 16 hidden columns
// (all three gates).  Global atomic barrier between the 64 time steps.
// ---------------------------------------------------------------------------
__global__ void __launch_bounds__(256)
gru_recurrence_kernel(const bf16_t* __restrict__ whh_bf,   // (2304,768)
                      const float*  __restrict__ b_hh,     // 2304
                      const float*  __restrict__ gie,      // (2048,2304) rows (t*32+b)
                      const float*  __restrict__ gimg,     // (32,2304), incl b_ih
                      const float*  __restrict__ h0_f32,   // (32,768)
                      const bf16_t* __restrict__ h0_bf,    // (32,768)
                      bf16_t* __restrict__ hall_bf,        // (2048,768)
                      float*  __restrict__ hall_f32,       // (64,32,768)
                      unsigned* __restrict__ counter,
                      int nblocks) {
    __shared__ float gh_s[3][BATCH][16];
    const int j0   = blockIdx.x * 16;
    const int lane = threadIdx.x & 31;
    const int wave = threadIdx.x >> 5;
    const int half = lane >> 4;
    const int r16  = lane & 15;

    for (int t = 0; t < SEQ; ++t) {
        const bf16_t* Ab = (t == 0) ? h0_bf
                                    : (hall_bf + (size_t)(t - 1) * BATCH * HID);
        if (wave < 6) {                       // wave-uniform: EXEC all-ones inside
            const int g  = wave >> 1;         // gate 0..2
            const int mh = wave & 1;          // batch half
            const bf16_t* Arow = Ab + (size_t)(mh * 16 + r16) * HID;
            const bf16_t* Brow = whh_bf + (size_t)(g * HID + j0 + r16) * HID;
            v8f acc = {};
            for (int kb = 0; kb < HID; kb += 32) {
                v16bf af = load_a_frag(Arow, kb, half);
                v16bf bfrag = *reinterpret_cast<const v16bf*>(Brow + kb + 16 * half);
                acc = __builtin_amdgcn_wmma_f32_16x16x32_bf16(
                    false, af, false, bfrag, (short)0, acc, false, false);
            }
#pragma unroll
            for (int v = 0; v < 8; ++v)
                gh_s[g][mh * 16 + half * 8 + v][r16] = acc[v];
        }
        __syncthreads();

        // elementwise gate combine: 512 (b,j) elements, 256 threads
        for (int e = threadIdx.x; e < BATCH * 16; e += 256) {
            int b  = e >> 4;
            int jl = e & 15;
            int j  = j0 + jl;
            size_t girow = ((size_t)t * BATCH + b) * G3;
            size_t gmrow = (size_t)b * G3;
            float i_r = gie[girow + j]           + gimg[gmrow + j];
            float i_z = gie[girow + HID + j]     + gimg[gmrow + HID + j];
            float i_n = gie[girow + 2 * HID + j] + gimg[gmrow + 2 * HID + j];
            float h_r = gh_s[0][b][jl] + b_hh[j];
            float h_z = gh_s[1][b][jl] + b_hh[HID + j];
            float h_n = gh_s[2][b][jl] + b_hh[2 * HID + j];
            float r = sigmoidf_(i_r + h_r);
            float z = sigmoidf_(i_z + h_z);
            float n = tanhf(i_n + r * h_n);
            float hp = (t == 0) ? h0_f32[(size_t)b * HID + j]
                                : hall_f32[((size_t)(t - 1) * BATCH + b) * HID + j];
            float hn = (1.0f - z) * n + z * hp;
            hall_f32[((size_t)t * BATCH + b) * HID + j] = hn;
            hall_bf [((size_t)t * BATCH + b) * HID + j] = (bf16_t)hn;
        }
        __syncthreads();

        // device-wide barrier between steps
        if (threadIdx.x == 0) {
            __threadfence();
            atomicAdd(counter, 1u);
            unsigned target = (unsigned)nblocks * (unsigned)(t + 1);
            while (atomicAdd(counter, 0u) < target) {
                __builtin_amdgcn_s_sleep(2);
            }
        }
        __syncthreads();
        __threadfence();   // acquire side: invalidate stale near caches
    }
}

__global__ void final_copy_kernel(const float* __restrict__ hall_f32,
                                  float* __restrict__ out_tail) {
    int i = blockIdx.x * blockDim.x + threadIdx.x;
    if (i < BATCH * HID)
        out_tail[i] = hall_f32[(size_t)(SEQ - 1) * BATCH * HID + i];
}

// ---------------------------------------------------------------------------
extern "C" void kernel_launch(void* const* d_in, const int* in_sizes, int n_in,
                              void* d_out, int out_size, void* d_ws, size_t ws_size,
                              hipStream_t stream) {
    (void)in_sizes; (void)n_in; (void)out_size; (void)ws_size;

    const int*   inputs = (const int*)  d_in[0];
    const float* hidden = (const float*)d_in[1];
    const float* memory = (const float*)d_in[2];
    const float* embed  = (const float*)d_in[3];
    const float* w_ih   = (const float*)d_in[4];
    const float* w_hh   = (const float*)d_in[5];
    const float* b_ih   = (const float*)d_in[6];
    const float* b_hh   = (const float*)d_in[7];
    const float* fc_w   = (const float*)d_in[8];
    const float* fc_b   = (const float*)d_in[9];
    float* out = (float*)d_out;

    char* ws = (char*)d_ws;
    size_t off = 0;
    auto alloc = [&](size_t bytes) -> char* {
        char* p = ws + off;
        off = (off + bytes + 255) & ~(size_t)255;
        return p;
    };
    bf16_t* fcw_bf  = (bf16_t*)alloc((size_t)VOCAB * HID * 2);
    bf16_t* wih_bf  = (bf16_t*)alloc((size_t)G3 * XDIM * 2);
    bf16_t* whh_bf  = (bf16_t*)alloc((size_t)G3 * HID * 2);
    bf16_t* emb_bf  = (bf16_t*)alloc((size_t)BATCH * SEQ * HID * 2);
    bf16_t* img_bf  = (bf16_t*)alloc((size_t)BATCH * HID * 2);
    bf16_t* h0_bf   = (bf16_t*)alloc((size_t)BATCH * HID * 2);
    float*  gie     = (float*) alloc((size_t)BATCH * SEQ * G3 * 4);
    float*  gimg    = (float*) alloc((size_t)BATCH * G3 * 4);
    bf16_t* hall_bf = (bf16_t*)alloc((size_t)BATCH * SEQ * HID * 2);
    float*  hall_f  = (float*) alloc((size_t)BATCH * SEQ * HID * 4);
    unsigned* counter = (unsigned*)alloc(256);

    auto cvt = [&](const float* s, bf16_t* d, size_t n) {
        f32_to_bf16_kernel<<<dim3((unsigned)((n + 255) / 256)), dim3(256), 0, stream>>>(s, d, n);
    };
    cvt(fc_w,   fcw_bf, (size_t)VOCAB * HID);
    cvt(w_ih,   wih_bf, (size_t)G3 * XDIM);
    cvt(w_hh,   whh_bf, (size_t)G3 * HID);
    cvt(memory, img_bf, (size_t)BATCH * HID);   // (B,1,HID) contiguous
    cvt(hidden, h0_bf,  (size_t)BATCH * HID);

    gather_emb_kernel<<<dim3((BATCH * SEQ * HID) / 256), dim3(256), 0, stream>>>(
        inputs, embed, emb_bf);
    zero_counter_kernel<<<dim3(1), dim3(64), 0, stream>>>(counter);

    // gie = emb @ w_ih[:, :HID]^T          (M=2048, N=2304, K=768)
    gemm_bf16_tn_blk_kernel<<<dim3(G3 / 128, (BATCH * SEQ) / 128), dim3(256), 0, stream>>>(
        emb_bf, HID, wih_bf, XDIM, nullptr, gie, G3, HID, 0);

    // gimg = img @ w_ih[:, HID:]^T + b_ih  (M=32, N=2304, K=768)
    gemm_bf16_tn_kernel<<<dim3(G3 / 64, 1), dim3(256), 0, stream>>>(
        img_bf, HID, wih_bf + HID, XDIM, b_ih, gimg, G3, HID);

    // sequential GRU recurrence: 48 WGs (768/16 columns), 64 steps
    gru_recurrence_kernel<<<dim3(HID / 16), dim3(256), 0, stream>>>(
        whh_bf, b_hh, gie, gimg, hidden, h0_bf, hall_bf, hall_f, counter, HID / 16);

    // logits = h_all @ fc_w^T + fc_b, permuted (b,t) store into d_out
    gemm_bf16_tn_blk_kernel<<<dim3(VOCAB / 128, (BATCH * SEQ) / 128), dim3(256), 0, stream>>>(
        hall_bf, HID, fcw_bf, HID, fc_b, out, VOCAB, HID, 1);

    // h_final tail
    final_copy_kernel<<<dim3((BATCH * HID) / 256), dim3(256), 0, stream>>>(
        hall_f, out + (size_t)BATCH * SEQ * VOCAB);
}
